// DeepMultipleDecoder_83382495085122
// MI455X (gfx1250) — matrix operand order repaired and checked by reading, not compiled, not run on hardware
//
#include <hip/hip_runtime.h>
#include <hip/hip_bf16.h>
#include <math.h>

// ---------------------------------------------------------------------------
// Problem constants (from reference): B=16, LP=128, LM=256, V=32000, E=512,
// H=512, DENC=1024.  All GEMMs have M = multiple of 16, N = multiple of 16,
// K = multiple of 32 -> exact WMMA 16x16x32 tiling, no edge handling needed.
// ---------------------------------------------------------------------------

typedef __attribute__((ext_vector_type(16))) _Float16 v16h;
typedef __attribute__((ext_vector_type(8)))  float    v8f;

// d_out layout (floats), in reference return order:
//  logits[16,32000], p_w[16,128,1], m_w[16,256,1], ctx_new[16,2048],
//  h1,c1,h2,c2,h3,c3,h4,c4 (each [16,512])
static constexpr int OFF_LOGITS = 0;
static constexpr int OFF_PW  = 512000;
static constexpr int OFF_MW  = 514048;
static constexpr int OFF_CTX = 518144;
static constexpr int OFF_H1  = 550912;
static constexpr int OFF_C1  = 559104;
static constexpr int OFF_H2  = 567296;
static constexpr int OFF_C2  = 575488;
static constexpr int OFF_H3  = 583680;
static constexpr int OFF_C3  = 591872;
static constexpr int OFF_H4  = 600064;
static constexpr int OFF_C4  = 608256;

// ---------------------------------------------------------------------------
// Core K-loop: accumulate A[16 rows, kbeg..kend) x W[kbeg..kend, 16 cols]
// into a WMMA f32 accumulator.  fp32 data in HBM (bandwidth-bound), converted
// to f16 fragments in registers (WMMA f16 throughput >> roofline need).
//
// Fragment layouts used here (wave32; A and B both use contiguous k order):
//   A 16x32 f16 : lane = half*16 + m ; v16h elem i = A[m][k0 + half*16 + i]
//   B 32x16 f16 : lane = half*16 + n ; v16h elem i = W[k0 + half*16 + i][n]
//   C 16x16 f32 : lane = half*16 + n ; v8f elem r -> row m = half*8 + r
//
// WNT: weights are single-pass in the LSTM/vocab GEMMs -> non-temporal loads
// so 327 MB of out_w does not evict reusable operands from WGP$/L2.
// ---------------------------------------------------------------------------
template <bool WNT>
__device__ __forceinline__ v8f wmma_kloop(const float* __restrict__ A, int lda,
                                          const float* __restrict__ W, int ldw,
                                          int row, int ncol, int half,
                                          int kbeg, int kend, v8f acc)
{
    // Per-lane base pointers; inner loop only does pointer increments.
    const float* ar = A + (size_t)row * lda + kbeg + half * 16;
    const float* wc = W + (size_t)(kbeg + half * 16) * ldw + ncol;
    const size_t wstep = (size_t)32 * ldw;

    for (int k0 = kbeg; k0 < kend; k0 += 32) {
        v16h a, b;
        float4 a0 = *reinterpret_cast<const float4*>(ar);
        float4 a1 = *reinterpret_cast<const float4*>(ar + 4);
        float4 a2 = *reinterpret_cast<const float4*>(ar + 8);
        float4 a3 = *reinterpret_cast<const float4*>(ar + 12);
        a[0]  = (_Float16)a0.x; a[1]  = (_Float16)a0.y;
        a[2]  = (_Float16)a0.z; a[3]  = (_Float16)a0.w;
        a[4]  = (_Float16)a1.x; a[5]  = (_Float16)a1.y;
        a[6]  = (_Float16)a1.z; a[7]  = (_Float16)a1.w;
        a[8]  = (_Float16)a2.x; a[9]  = (_Float16)a2.y;
        a[10] = (_Float16)a2.z; a[11] = (_Float16)a2.w;
        a[12] = (_Float16)a3.x; a[13] = (_Float16)a3.y;
        a[14] = (_Float16)a3.z; a[15] = (_Float16)a3.w;

#pragma unroll
        for (int i = 0; i < 16; ++i) {
            float w = WNT ? __builtin_nontemporal_load(wc + (size_t)i * ldw)
                          : wc[(size_t)i * ldw];
            b[i] = (_Float16)w;
        }

        acc = __builtin_amdgcn_wmma_f32_16x16x32_f16(
            /*neg_a=*/false, a, /*neg_b=*/false, b,
            /*c_mod=*/(short)0, acc, /*reuse_a=*/false, /*reuse_b=*/false);

        ar += 32;
        wc += wstep;
    }
    return acc;
}

// Generic WMMA GEMM:  C[M,N] (+)= A[M,K]@W[K,N] [+ A2[M,K2]@W2[K2,N]] + bias
// grid = (N/16, M/16, ksplit); block = 32 (one wave per 16x16 tile).
// ksplit==1  -> plain store.  ksplit>1 -> hardware f32 atomic accumulate
// (C pre-zeroed); bias and the optional second operand pair come from z==0.
template <bool WNT>
__global__ void k_wmma_gemm(const float* __restrict__ A, int lda,
                            const float* __restrict__ W,
                            const float* __restrict__ A2,
                            const float* __restrict__ W2, int K2,
                            const float* __restrict__ bias,
                            float* __restrict__ C, int N, int K)
{
    const int lane = threadIdx.x & 31;
    const int half = lane >> 4;
    const int lm   = lane & 15;
    const int row  = blockIdx.y * 16 + lm;   // A row handled by this lane
    const int ncol = blockIdx.x * 16 + lm;   // C/B column handled by this lane

    const int nz  = (int)gridDim.z;
    const int kc  = K / nz;                  // K multiple of 32*nz by design
    const int kb  = (int)blockIdx.z * kc;
    const int ke  = ((int)blockIdx.z == nz - 1) ? K : kb + kc;

    v8f acc = {};
    acc = wmma_kloop<WNT>(A, lda, W, N, row, ncol, half, kb, ke, acc);
    if (A2 != nullptr && blockIdx.z == 0)
        acc = wmma_kloop<false>(A2, K2, W2, N, row, ncol, half, 0, K2, acc);

    const float bv = (bias != nullptr && blockIdx.z == 0) ? bias[ncol] : 0.0f;
#pragma unroll
    for (int r = 0; r < 8; ++r) {
        const int m = half * 8 + r;
        float* p = &C[(size_t)(blockIdx.y * 16 + m) * N + ncol];
        const float v = acc[r] + bv;
        if (nz == 1) {
            *p = v;
        } else {
            // relaxed agent-scope fadd -> global_atomic_add_f32 (no CAS loop)
            __hip_atomic_fetch_add(p, v, __ATOMIC_RELAXED,
                                   __HIP_MEMORY_SCOPE_AGENT);
        }
    }
}

// ---------------------------------------------------------------------------
// Elementwise / reduction kernels
// ---------------------------------------------------------------------------
__device__ __forceinline__ float sigf(float x) { return 1.0f / (1.0f + __expf(-x)); }

// x = concat(ctx[16,2048], emb_table[word][16,512]) -> ws_x[16,2560]
__global__ void k_build_x(const int* __restrict__ word,
                          const float* __restrict__ ctx,
                          const float* __restrict__ emb,
                          float* __restrict__ x)
{
    int idx = blockIdx.x * blockDim.x + threadIdx.x;
    if (idx >= 16 * 2560) return;
    int b = idx / 2560, j = idx - b * 2560;
    float v = (j < 2048) ? ctx[b * 2048 + j]
                         : emb[(size_t)word[b] * 512 + (j - 2048)];
    x[idx] = v;
}

// z[16,2048] (i|f|g|o) + c_prev -> h_new, c_new
__global__ void k_lstm_gates(const float* __restrict__ z,
                             const float* __restrict__ c_prev,
                             float* __restrict__ h_out,
                             float* __restrict__ c_out)
{
    int idx = blockIdx.x * blockDim.x + threadIdx.x;
    if (idx >= 16 * 512) return;
    int b = idx >> 9, j = idx & 511;
    const float* zr = z + (size_t)b * 2048;
    float zi = zr[j], zf = zr[512 + j], zg = zr[1024 + j], zo = zr[1536 + j];
    float c = sigf(zf) * c_prev[idx] + sigf(zi) * tanhf(zg);
    float h = sigf(zo) * tanhf(c);
    c_out[idx] = c;
    h_out[idx] = h;
}

// score[b,l] = tanh(S1[b*L+l,:] + t2[b,:]) . Vw + Vb   (one wave32 per row)
__global__ void k_score(const float* __restrict__ S1,
                        const float* __restrict__ t2,
                        const float* __restrict__ Vw,
                        const float* __restrict__ Vb,
                        int L, float* __restrict__ out)
{
    int wid  = (blockIdx.x * blockDim.x + threadIdx.x) >> 5;
    int lane = threadIdx.x & 31;
    if (wid >= 16 * L) return;
    int b = wid / L;
    const float* s1 = S1 + (size_t)wid * 512;
    const float* tt = t2 + (size_t)b * 512;
    float s = 0.0f;
    for (int n = lane; n < 512; n += 32)
        s += tanhf(s1[n] + tt[n]) * Vw[n];
#pragma unroll
    for (int off = 16; off; off >>= 1) s += __shfl_xor(s, off, 32);
    if (lane == 0) out[wid] = s + Vb[0];
}

// in-place softmax over L (L = 128 or 256, power of two; block = L threads)
__global__ void k_softmax(float* __restrict__ w, int L)
{
    __shared__ float sm[256];
    int b = blockIdx.x, t = threadIdx.x;
    float* row = w + (size_t)b * L;
    float v = row[t];
    sm[t] = v; __syncthreads();
    for (int s = L >> 1; s; s >>= 1) {
        if (t < s) sm[t] = fmaxf(sm[t], sm[t + s]);
        __syncthreads();
    }
    float mx = sm[0]; __syncthreads();
    float e = __expf(v - mx);
    sm[t] = e; __syncthreads();
    for (int s = L >> 1; s; s >>= 1) {
        if (t < s) sm[t] += sm[t + s];
        __syncthreads();
    }
    row[t] = e / sm[0];
}

// ctx[b, off+d] = sum_l w[b,l] * enc[b,l,d]   (d in 0..1023)
__global__ void k_ctx(const float* __restrict__ w,
                      const float* __restrict__ enc,
                      int L, float* __restrict__ ctx_out, int off)
{
    int b = blockIdx.x;
    int d = blockIdx.y * blockDim.x + threadIdx.x;
    const float* e  = enc + (size_t)b * L * 1024 + d;
    const float* wr = w + (size_t)b * L;
    float s = 0.0f;
    for (int l = 0; l < L; ++l) s += wr[l] * e[(size_t)l * 1024];
    ctx_out[(size_t)b * 2048 + off + d] = s;
}

// xo = concat(h4[16,512], ctx_new[16,2048]) -> ws_xo[16,2560]
__global__ void k_build_xo(const float* __restrict__ h4,
                           const float* __restrict__ ctxn,
                           float* __restrict__ xo)
{
    int idx = blockIdx.x * blockDim.x + threadIdx.x;
    if (idx >= 16 * 2560) return;
    int b = idx / 2560, j = idx - b * 2560;
    xo[idx] = (j < 512) ? h4[b * 512 + j] : ctxn[(size_t)b * 2048 + (j - 512)];
}

__global__ void k_zero(float* __restrict__ p, int n)
{
    int i = blockIdx.x * blockDim.x + threadIdx.x;
    if (i < n) p[i] = 0.0f;
}

// ---------------------------------------------------------------------------
// Host launcher
// ---------------------------------------------------------------------------
extern "C" void kernel_launch(void* const* d_in, const int* in_sizes, int n_in,
                              void* d_out, int out_size, void* d_ws, size_t ws_size,
                              hipStream_t stream)
{
    (void)in_sizes; (void)n_in; (void)out_size; (void)ws_size;

    const int*   word   = (const int*)  d_in[0];
    const float* encP   = (const float*)d_in[1];
    const float* encM   = (const float*)d_in[2];
    const float* ctx_in = (const float*)d_in[3];
    const float* h_in[4] = {(const float*)d_in[4], (const float*)d_in[6],
                            (const float*)d_in[8], (const float*)d_in[10]};
    const float* c_in[4] = {(const float*)d_in[5], (const float*)d_in[7],
                            (const float*)d_in[9], (const float*)d_in[11]};
    const float* emb    = (const float*)d_in[12];
    const float* Wx[4]  = {(const float*)d_in[13], (const float*)d_in[16],
                           (const float*)d_in[19], (const float*)d_in[22]};
    const float* Wh[4]  = {(const float*)d_in[14], (const float*)d_in[17],
                           (const float*)d_in[20], (const float*)d_in[23]};
    const float* bl[4]  = {(const float*)d_in[15], (const float*)d_in[18],
                           (const float*)d_in[21], (const float*)d_in[24]};
    const float* pW1w = (const float*)d_in[25]; const float* pW1b = (const float*)d_in[26];
    const float* pW2w = (const float*)d_in[27]; const float* pW2b = (const float*)d_in[28];
    const float* pVw  = (const float*)d_in[29]; const float* pVb  = (const float*)d_in[30];
    const float* mW1w = (const float*)d_in[31]; const float* mW1b = (const float*)d_in[32];
    const float* mW2w = (const float*)d_in[33]; const float* mW2b = (const float*)d_in[34];
    const float* mVw  = (const float*)d_in[35]; const float* mVb  = (const float*)d_in[36];
    const float* outW = (const float*)d_in[37]; const float* outB = (const float*)d_in[38];

    float* out = (float*)d_out;
    float* ws  = (float*)d_ws;

    // workspace layout (floats)
    float* ws_x   = ws;                 // 16*2560
    float* ws_xo  = ws_x  + 40960;      // 16*2560
    float* ws_z   = ws_xo + 40960;      // 16*2048
    float* ws_t2p = ws_z   + 32768;     // 16*512
    float* ws_t2m = ws_t2p + 8192;      // 16*512
    float* ws_S1p = ws_t2m + 8192;      // 2048*512
    float* ws_S1m = ws_S1p + 1048576;   // 4096*512

    float* h_out[4] = {out + OFF_H1, out + OFF_H2, out + OFF_H3, out + OFF_H4};
    float* c_out[4] = {out + OFF_C1, out + OFF_C2, out + OFF_C3, out + OFF_C4};

    // 1) x = [ctx, emb[word]]
    k_build_x<<<160, 256, 0, stream>>>(word, ctx_in, emb, ws_x);

    // 2) four sequential LSTM layers: z = x@Wx + h@Wh + b, then gates
    //    (weights are single-pass -> non-temporal)
    const float* lx   = ws_x;
    int          lxK  = 2560;
    for (int li = 0; li < 4; ++li) {
        k_wmma_gemm<true><<<dim3(128, 1, 1), 32, 0, stream>>>(
            lx, lxK, Wx[li], h_in[li], Wh[li], 512, bl[li], ws_z, 2048, lxK);
        k_lstm_gates<<<32, 256, 0, stream>>>(ws_z, c_in[li], h_out[li], c_out[li]);
        lx  = h_out[li];
        lxK = 512;
    }
    const float* h4 = h_out[3];

    // 3) attention precomputes: t2 = h4@W2 + b2 ; S1 = enc@W1 + b1
    //    (W1/W2 are re-read by many M-tiles -> keep regular temporal)
    k_wmma_gemm<false><<<dim3(32, 1, 1), 32, 0, stream>>>(
        h4, 512, pW2w, nullptr, nullptr, 0, pW2b, ws_t2p, 512, 512);
    k_wmma_gemm<false><<<dim3(32, 1, 1), 32, 0, stream>>>(
        h4, 512, mW2w, nullptr, nullptr, 0, mW2b, ws_t2m, 512, 512);
    k_wmma_gemm<false><<<dim3(32, 128, 1), 32, 0, stream>>>(
        encP, 1024, pW1w, nullptr, nullptr, 0, pW1b, ws_S1p, 512, 1024);
    k_wmma_gemm<false><<<dim3(32, 256, 1), 32, 0, stream>>>(
        encM, 1024, mW1w, nullptr, nullptr, 0, mW1b, ws_S1m, 512, 1024);

    // 4) scores (tanh-dot, wave32 reduction) written into d_out weight slots
    k_score<<<512, 128, 0, stream>>>(ws_S1p, ws_t2p, pVw, pVb, 128, out + OFF_PW);
    k_score<<<1024, 128, 0, stream>>>(ws_S1m, ws_t2m, mVw, mVb, 256, out + OFF_MW);

    // 5) softmax in place
    k_softmax<<<16, 128, 0, stream>>>(out + OFF_PW, 128);
    k_softmax<<<16, 256, 0, stream>>>(out + OFF_MW, 256);

    // 6) ctx_new = [sum w_p*encP , sum w_m*encM]
    k_ctx<<<dim3(16, 4, 1), 256, 0, stream>>>(out + OFF_PW, encP, 128, out + OFF_CTX, 0);
    k_ctx<<<dim3(16, 4, 1), 256, 0, stream>>>(out + OFF_MW, encM, 256, out + OFF_CTX, 1024);

    // 7) logits = [h4, ctx_new] @ out_w + out_b   (split-K x4, HW f32 atomics,
    //    non-temporal weight stream: out_w is 327 MB read exactly once)
    k_build_xo<<<160, 256, 0, stream>>>(h4, out + OFF_CTX, ws_xo);
    k_zero<<<2000, 256, 0, stream>>>(out + OFF_LOGITS, 16 * 32000);
    k_wmma_gemm<true><<<dim3(2000, 1, 4), 32, 0, stream>>>(
        ws_xo, 2560, outW, nullptr, nullptr, 0, outB, out + OFF_LOGITS, 32000, 2560);
}
